// transformer_43920335569209
// MI455X (gfx1250) — matrix operand order repaired and checked by reading, not compiled, not run on hardware
//
#include <hip/hip_runtime.h>

typedef __attribute__((ext_vector_type(16))) _Float16 v16h;
typedef __attribute__((ext_vector_type(8)))  _Float16 v8h;
typedef __attribute__((ext_vector_type(4)))  _Float16 v4h;
typedef __attribute__((ext_vector_type(8)))  float    v8f;
typedef __attribute__((ext_vector_type(8)))  unsigned int v8u;

#define BB 16
#define CC 128
#define NN 2048
#define CQD 32   // C/4 = head dim = WMMA K

// ---------------- 1x1 conv + BN(eval) + ReLU, f32 math, f16 output -------
template<int OC, bool TR>
__global__ __launch_bounds__(256) void proj_kernel(
    const float* __restrict__ x, const float* __restrict__ w,
    const float* __restrict__ bias, const float* __restrict__ s,
    const float* __restrict__ t, _Float16* __restrict__ out)
{
  const int n = blockIdx.x * 256 + threadIdx.x;
  const int b = blockIdx.y;
  const float* xb = x + (size_t)b * CC * NN + n;
  for (int o0 = 0; o0 < OC; o0 += 16) {
    float acc[16];
#pragma unroll
    for (int j = 0; j < 16; ++j) acc[j] = bias[o0 + j];
    for (int c = 0; c < CC; ++c) {
      const float xv = xb[(size_t)c * NN];
#pragma unroll
      for (int j = 0; j < 16; ++j)
        acc[j] = fmaf(w[(o0 + j) * CC + c], xv, acc[j]);
    }
#pragma unroll
    for (int j = 0; j < 16; ++j) {
      const int o = o0 + j;
      const float r = fmaxf(0.f, fmaf(s[o], acc[j], t[o]));
      if (TR) out[((size_t)b * NN + n) * OC + o] = (_Float16)r;
      else    out[((size_t)b * OC + o) * NN + n] = (_Float16)r;
    }
  }
}

// ---------------- flash attention: Sᵀ = kᵀ×q, online softmax, Oᵀ = v×Pᵀ ----
// grid: (N/128, B), 256 threads = 8 waves; wave w owns n-tile [nblk*128+16w, +16)
// Double-buffered LDS staging so next chunk's global loads overlap compute.
__global__ __launch_bounds__(256) void flash_attn_kernel(
    const _Float16* __restrict__ qT,   // [B][N][32]
    const _Float16* __restrict__ kT,   // [B][N][32]
    const _Float16* __restrict__ vsrc, // [B][128][N]
    _Float16* __restrict__ attn)       // [B][128][N]
{
  __shared__ __attribute__((aligned(16))) _Float16 lk[2][32][40];   // [m_local][d]
  __shared__ __attribute__((aligned(16))) _Float16 lv[2][128][40];  // [c][m_local]

  const int b    = blockIdx.y;
  const int nblk = blockIdx.x;
  const int tid  = threadIdx.x;
  const int wave = tid >> 5;
  const int lane = tid & 31;
  const int lr   = lane & 15;
  const bool hif = lane >= 16;
  const int d0   = hif ? 16 : 0;         // A/B fragment K-half select
  const int ncol = nblk * 128 + wave * 16 + lr;

  // q B-fragment (K=d ascending per lane half): resident for whole kernel
  const _Float16* qrow = qT + ((size_t)b * NN + ncol) * CQD + (hif ? 16 : 0);
  const v16h qf = *(const v16h*)qrow;

  const _Float16* kTb = kT   + (size_t)b * NN * CQD;
  const _Float16* vb  = vsrc + (size_t)b * CC * NN;

  // per-thread staging coordinates
  const int kml = tid >> 3, kdd = (tid & 7) * 4;      // k chunk: 4 halfs/thread
  const int vc  = tid >> 1, vmh = (tid & 1) * 16;     // v chunk: 16 halfs/thread

  v8f acc[8];
#pragma unroll
  for (int ct = 0; ct < 8; ++ct)
#pragma unroll
    for (int j = 0; j < 8; ++j) acc[ct][j] = 0.f;

  float m_run = -__builtin_inff();
  float l_run = 0.f;

  // prologue: stage chunk 0 into buffer 0
  {
    *(v4h*)&lk[0][kml][kdd] = *(const v4h*)(kTb + (size_t)kml * CQD + kdd);
    const v8h* src = (const v8h*)(vb + (size_t)vc * NN + vmh);
    *(v8h*)&lv[0][vc][vmh]     = src[0];
    *(v8h*)&lv[0][vc][vmh + 8] = src[1];
  }

  for (int it = 0; it < NN / 32; ++it) {
    const int m0  = it * 32;
    const int buf = it & 1;
    __syncthreads();   // staged buf visible; prior reads of buf^1 complete

    // stage next chunk into the other buffer (overlaps with compute below)
    if (m0 + 32 < NN) {
      const int nb = buf ^ 1;
      *(v4h*)&lk[nb][kml][kdd] =
          *(const v4h*)(kTb + (size_t)(m0 + 32 + kml) * CQD + kdd);
      const v8h* src = (const v8h*)(vb + (size_t)vc * NN + m0 + 32 + vmh);
      *(v8h*)&lv[nb][vc][vmh]     = src[0];
      *(v8h*)&lv[nb][vc][vmh + 8] = src[1];
    }

    // kᵀ A-fragments for two 16-row m tiles
    v16h kf0, kf1;
    {
      const v8h a0 = *(const v8h*)&lk[buf][lr][d0];
      const v8h a1 = *(const v8h*)&lk[buf][lr][d0 + 8];
      const v8h b0 = *(const v8h*)&lk[buf][16 + lr][d0];
      const v8h b1 = *(const v8h*)&lk[buf][16 + lr][d0 + 8];
#pragma unroll
      for (int j = 0; j < 8; ++j) {
        kf0[j] = a0[j]; kf0[8 + j] = a1[j];
        kf1[j] = b0[j]; kf1[8 + j] = b1[j];
      }
    }

    v8f z;
#pragma unroll
    for (int j = 0; j < 8; ++j) z[j] = 0.f;

    // Sᵀ tiles: rows m, cols n (lane = n, VGPR = m)
    v8f st0 = __builtin_amdgcn_wmma_f32_16x16x32_f16(false, kf0, false, qf,
                                                     (short)0, z, false, false);
    v8f st1 = __builtin_amdgcn_wmma_f32_16x16x32_f16(false, kf1, false, qf,
                                                     (short)0, z, false, false);

    // online softmax over this 32-wide m chunk (per n column)
    float mloc = st0[0];
#pragma unroll
    for (int j = 1; j < 8; ++j) mloc = fmaxf(mloc, st0[j]);
#pragma unroll
    for (int j = 0; j < 8; ++j) mloc = fmaxf(mloc, st1[j]);
    const float mchunk = fmaxf(mloc, __shfl_xor(mloc, 16));
    const float mnew   = fmaxf(m_run, mchunk);
    const float corr   = __expf(m_run - mnew);

    float p0[8], p1[8], ls = 0.f;
#pragma unroll
    for (int j = 0; j < 8; ++j) { p0[j] = __expf(st0[j] - mnew); ls += p0[j]; }
#pragma unroll
    for (int j = 0; j < 8; ++j) { p1[j] = __expf(st1[j] - mnew); ls += p1[j]; }
    l_run = l_run * corr + ls + __shfl_xor(ls, 16);
    m_run = mnew;
#pragma unroll
    for (int ct = 0; ct < 8; ++ct)
#pragma unroll
      for (int j = 0; j < 8; ++j) acc[ct][j] *= corr;

    // Pᵀ B-fragment: pack to f16 pairs first, then one half-swap shuffle per
    // packed u32 (8 bpermutes instead of 16)
    unsigned int pk0[4], pk1[4];
#pragma unroll
    for (int k = 0; k < 4; ++k) {
      pk0[k] = __builtin_bit_cast(unsigned int,
               __builtin_amdgcn_cvt_pkrtz(p0[2 * k], p0[2 * k + 1]));
      pk1[k] = __builtin_bit_cast(unsigned int,
               __builtin_amdgcn_cvt_pkrtz(p1[2 * k], p1[2 * k + 1]));
    }
    v8u pw;
#pragma unroll
    for (int k = 0; k < 4; ++k) {
      const unsigned int o0 = (unsigned int)__shfl_xor((int)pk0[k], 16);
      const unsigned int o1 = (unsigned int)__shfl_xor((int)pk1[k], 16);
      pw[k]     = hif ? o1     : pk0[k];
      pw[4 + k] = hif ? pk1[k] : o0;
    }
    const v16h pf = __builtin_bit_cast(v16h, pw);

    // Oᵀ += v(c×m) × Pᵀ(m×n), 8 c-tiles
#pragma unroll
    for (int ct = 0; ct < 8; ++ct) {
      const int c = ct * 16 + lr;
      v16h vf;
      const v8h a0 = *(const v8h*)&lv[buf][c][d0];
      const v8h a1 = *(const v8h*)&lv[buf][c][d0 + 8];
#pragma unroll
      for (int j = 0; j < 8; ++j) { vf[j] = a0[j]; vf[8 + j] = a1[j]; }
      acc[ct] = __builtin_amdgcn_wmma_f32_16x16x32_f16(false, vf, false, pf,
                                                       (short)0, acc[ct],
                                                       false, false);
    }
  }

  // normalize and emit attn[b][c][ncol]
  const float invl = 1.0f / l_run;
#pragma unroll
  for (int ct = 0; ct < 8; ++ct)
#pragma unroll
    for (int j = 0; j < 8; ++j) {
      const int c = ct * 16 + (hif ? 8 : 0) + j;
      attn[((size_t)b * CC + c) * NN + ncol] = (_Float16)(acc[ct][j] * invl);
    }
}

// ---------------- o-proj + BN + ReLU + residual ----------------
__global__ __launch_bounds__(256) void oproj_kernel(
    const _Float16* __restrict__ attn, const float* __restrict__ x_row,
    const float* __restrict__ wo, const float* __restrict__ bo,
    const float* __restrict__ so, const float* __restrict__ to,
    const float* __restrict__ gamma, float* __restrict__ out)
{
  const int n = blockIdx.x * 256 + threadIdx.x;
  const int b = blockIdx.y;
  const _Float16* ab = attn + (size_t)b * CC * NN + n;
  const float g = gamma[0];
  for (int o0 = 0; o0 < CC; o0 += 16) {
    float acc[16];
#pragma unroll
    for (int j = 0; j < 16; ++j) acc[j] = bo[o0 + j];
    for (int c = 0; c < CC; ++c) {
      const float a = (float)ab[(size_t)c * NN];
#pragma unroll
      for (int j = 0; j < 16; ++j)
        acc[j] = fmaf(wo[(o0 + j) * CC + c], a, acc[j]);
    }
#pragma unroll
    for (int j = 0; j < 16; ++j) {
      const int o = o0 + j;
      const float r = fmaxf(0.f, fmaf(so[o], acc[j], to[o]));
      const size_t idx = ((size_t)b * CC + o) * NN + n;
      out[idx] = fmaf(g, r, x_row[idx]);
    }
  }
}

extern "C" void kernel_launch(void* const* d_in, const int* in_sizes, int n_in,
                              void* d_out, int out_size, void* d_ws, size_t ws_size,
                              hipStream_t stream) {
  const float* x_row = (const float*)d_in[0];
  const float* x_loc = (const float*)d_in[1];
  const float* wq = (const float*)d_in[2];
  const float* bq = (const float*)d_in[3];
  const float* sq = (const float*)d_in[4];
  const float* tq = (const float*)d_in[5];
  const float* wk = (const float*)d_in[6];
  const float* bk = (const float*)d_in[7];
  const float* sk = (const float*)d_in[8];
  const float* tk = (const float*)d_in[9];
  const float* wv = (const float*)d_in[10];
  const float* bv = (const float*)d_in[11];
  const float* sv = (const float*)d_in[12];
  const float* tv = (const float*)d_in[13];
  const float* wo = (const float*)d_in[14];
  const float* bo = (const float*)d_in[15];
  const float* so = (const float*)d_in[16];
  const float* to = (const float*)d_in[17];
  const float* gm = (const float*)d_in[18];

  _Float16* qT = (_Float16*)d_ws;                       // [B][N][32]
  _Float16* kT = qT + (size_t)BB * NN * CQD;            // [B][N][32]
  _Float16* vb = kT + (size_t)BB * NN * CQD;            // [B][128][N]
  _Float16* at = vb + (size_t)BB * CC * NN;             // [B][128][N]

  dim3 pg(NN / 256, BB), blk(256);
  proj_kernel<CQD, true ><<<pg, blk, 0, stream>>>(x_loc, wq, bq, sq, tq, qT);
  proj_kernel<CQD, true ><<<pg, blk, 0, stream>>>(x_loc, wk, bk, sk, tk, kT);
  proj_kernel<CC,  false><<<pg, blk, 0, stream>>>(x_loc, wv, bv, sv, tv, vb);

  flash_attn_kernel<<<dim3(NN / 128, BB), blk, 0, stream>>>(qT, kT, vb, at);

  oproj_kernel<<<pg, blk, 0, stream>>>(at, x_row, wo, bo, so, to, gm,
                                       (float*)d_out);
}
